// AttentionLayer_86629490360750
// MI455X (gfx1250) — matrix operand order repaired and checked by reading, not compiled, not run on hardware
//
#include <hip/hip_runtime.h>
#include <math.h>

typedef __attribute__((ext_vector_type(16))) _Float16 v16h;
typedef __attribute__((ext_vector_type(8)))  _Float16 v8h;
typedef __attribute__((ext_vector_type(8)))  float    v8f;

#define B_    4
#define SQ_   4096
#define SKV_  4096
#define D_    64
#define TQ    64       // q rows per block (4 waves * 16)
#define TK    64       // kv rows per tile
#define NWAVES 4
#define NTHREADS (NWAVES * 32)
#define LPAD  8        // LDS row padding (halves) to dodge bank conflicts

#define LOG2E 1.44269504088896340736f

__device__ __forceinline__ v16h load16h(const _Float16* p) {
  // two 16B LDS loads (row bases are 16B aligned, not 32B)
  union { v16h v; v8h h[2]; } u;
  u.h[0] = *(const v8h*)(p);
  u.h[1] = *(const v8h*)(p + 8);
  return u.v;
}

__device__ __forceinline__ v8f wmma_f16(v16h a, v16h b, v8f c) {
  return __builtin_amdgcn_wmma_f32_16x16x32_f16(
      /*neg_a=*/false, a, /*neg_b=*/false, b,
      /*c_mod=*/(short)0, c, /*reuse_a=*/false, /*reuse_b=*/false);
}

__global__ __launch_bounds__(NTHREADS, 1)
void fa_fwd(const float* __restrict__ Q, const float* __restrict__ V,
            float* __restrict__ Out) {
  __shared__ _Float16 lds_v [TK][D_ + LPAD];   // V tile, row-major  [key][d]
  __shared__ _Float16 lds_vt[D_][TK + LPAD];   // V tile, transposed [d][key]
  __shared__ _Float16 lds_p [NWAVES][16][TK + LPAD]; // per-wave P bounce

  const int tid  = threadIdx.x;
  const int lane = tid & 31;
  const int wave = tid >> 5;
  const int g    = lane >> 4;   // 16-lane half-group
  const int ln   = lane & 15;

  const int blocksPerBatch = SQ_ / TQ;              // 64
  const int b     = blockIdx.x / blocksPerBatch;
  const int qt    = blockIdx.x % blocksPerBatch;
  const int qRow0 = qt * TQ + wave * 16;            // first q row of this wave

  const float* Qb = Q   + (size_t)b * SQ_  * D_;
  const float* Vb = V   + (size_t)b * SKV_ * D_;
  float*       Ob = Out + (size_t)b * SQ_  * D_;

  // ---- resident Q A-fragments, pre-scaled by log2(e) (softmax in log2 domain)
  // A layout: row M = lane%16, half h -> K = 16*g + h
  v16h aq0, aq1;
  {
    const float* qp = Qb + (size_t)(qRow0 + ln) * D_;
    #pragma unroll
    for (int h = 0; h < 16; ++h) {
      int d = (g << 4) + h;
      aq0[h] = (_Float16)(qp[d]      * LOG2E);
      aq1[h] = (_Float16)(qp[d + 32] * LOG2E);
    }
  }

  v8f vzero;
  #pragma unroll
  for (int i = 0; i < 8; ++i) vzero[i] = 0.0f;
  v16h vones;
  #pragma unroll
  for (int i = 0; i < 16; ++i) vones[i] = (_Float16)1.0f;

  v8f   oacc[4];                 // O tile 16x64 (fp32), 4 column sub-tiles
  float m[8], l[8];
  #pragma unroll
  for (int t = 0; t < 4; ++t) oacc[t] = vzero;
  #pragma unroll
  for (int r = 0; r < 8; ++r) { m[r] = -3.0e38f; l[r] = 0.0f; }

  // ---- software pipeline: prefetch V tile kv=0 into registers
  // 128 threads cover 64x64 fp32: 2 threads per key row, 32 d each
  const int vrow  = tid >> 1;           // 0..63  (key within tile)
  const int vcol0 = (tid & 1) << 5;     // 0 or 32 (d)
  float4 pre[8];
  {
    const float* vp = Vb + (size_t)vrow * D_ + vcol0;
    #pragma unroll
    for (int j = 0; j < 8; ++j) pre[j] = *(const float4*)(vp + 4 * j);
  }

  for (int kv = 0; kv < SKV_; kv += TK) {
    __syncthreads();   // previous tile fully consumed
    // ---- commit prefetched tile to LDS as f16, both orientations
    #pragma unroll
    for (int j = 0; j < 8; ++j) {
      int c = vcol0 + 4 * j;
      _Float16 h0 = (_Float16)pre[j].x, h1 = (_Float16)pre[j].y,
               h2 = (_Float16)pre[j].z, h3 = (_Float16)pre[j].w;
      lds_v [vrow][c + 0] = h0;  lds_v [vrow][c + 1] = h1;
      lds_v [vrow][c + 2] = h2;  lds_v [vrow][c + 3] = h3;
      lds_vt[c + 0][vrow] = h0;  lds_vt[c + 1][vrow] = h1;
      lds_vt[c + 2][vrow] = h2;  lds_vt[c + 3][vrow] = h3;
    }
    __syncthreads();

    // ---- issue global loads for the NEXT tile; they retire under the WMMAs
    if (kv + TK < SKV_) {
      const float* vp = Vb + (size_t)(kv + TK + vrow) * D_ + vcol0;
      #pragma unroll
      for (int j = 0; j < 8; ++j) pre[j] = *(const float4*)(vp + 4 * j);
    }

    // ---- S(16x64) = Q_tile x V_tile^T ; B[k=d][n=key]
    // B layout: N = lane%16, half h -> K = 16*g + h  => read lds_v[key][d..]
    v8f sacc[4];
    #pragma unroll
    for (int t = 0; t < 4; ++t) {
      v16h b0 = load16h(&lds_v[16 * t + ln][16 * g]);        // d in [0,32)
      v16h b1 = load16h(&lds_v[16 * t + ln][32 + 16 * g]);   // d in [32,64)
      v8f s = vzero;
      s = wmma_f16(aq0, b0, s);
      s = wmma_f16(aq1, b1, s);
      sacc[t] = s;
    }

    // ---- online softmax (log2 domain); row M = r + 8*g lives in C-vgpr r
    float scale[8];
    #pragma unroll
    for (int r = 0; r < 8; ++r) {
      float v0 = fmaxf(fmaxf(sacc[0][r], sacc[1][r]),
                       fmaxf(sacc[2][r], sacc[3][r]));
      #pragma unroll
      for (int mk = 1; mk < 16; mk <<= 1)
        v0 = fmaxf(v0, __shfl_xor(v0, mk, 16));
      float mn = fmaxf(m[r], v0);
      scale[r] = exp2f(m[r] - mn);
      m[r] = mn;
    }

    #pragma unroll
    for (int t = 0; t < 4; ++t) {
      #pragma unroll
      for (int r = 0; r < 8; ++r)
        sacc[t][r] = exp2f(sacc[t][r] - m[r]);   // P in [0,1], f16-safe
    }

    // ---- P: C-layout -> per-wave LDS -> A-fragment layout
    #pragma unroll
    for (int t = 0; t < 4; ++t) {
      #pragma unroll
      for (int r = 0; r < 8; ++r)
        lds_p[wave][r + 8 * g][16 * t + ln] = (_Float16)sacc[t][r];
    }
    asm volatile("s_wait_dscnt 0x0" ::: "memory");

    v16h pa0, pa1;
    {
      union { v16h v; v8h h[2]; } u0, u1;
      u0.h[0] = *(const v8h*)&lds_p[wave][ln][     16 * g];   // keys [0,32)
      u0.h[1] = *(const v8h*)&lds_p[wave][ln][8 +  16 * g];
      u1.h[0] = *(const v8h*)&lds_p[wave][ln][32 + 16 * g];   // keys [32,64)
      u1.h[1] = *(const v8h*)&lds_p[wave][ln][40 + 16 * g];
      pa0 = u0.v; pa1 = u1.v;
    }

    // ---- row sums via the matrix pipe: rowsum(P) = P x ones (C-layout, already
    //      replicated across the 16 columns of each half-group)
    v8f rsum = vzero;
    rsum = wmma_f16(pa0, vones, rsum);
    rsum = wmma_f16(pa1, vones, rsum);
    #pragma unroll
    for (int r = 0; r < 8; ++r)
      l[r] = l[r] * scale[r] + rsum[r];

    #pragma unroll
    for (int t = 0; t < 4; ++t) {
      #pragma unroll
      for (int r = 0; r < 8; ++r)
        oacc[t][r] *= scale[r];
    }

    // ---- O += P x V_tile ; B[k=key][n=d] => read lds_vt[d][key..]
    #pragma unroll
    for (int t = 0; t < 4; ++t) {
      v16h b0 = load16h(&lds_vt[16 * t + ln][16 * g]);        // keys [0,32)
      v16h b1 = load16h(&lds_vt[16 * t + ln][32 + 16 * g]);   // keys [32,64)
      oacc[t] = wmma_f16(pa0, b0, oacc[t]);
      oacc[t] = wmma_f16(pa1, b1, oacc[t]);
    }
  }

  // ---- normalize and write out (coalesced fp32)
  float inv[8];
  #pragma unroll
  for (int r = 0; r < 8; ++r) inv[r] = 1.0f / l[r];
  #pragma unroll
  for (int t = 0; t < 4; ++t) {
    #pragma unroll
    for (int r = 0; r < 8; ++r) {
      int qRow = qRow0 + r + 8 * g;
      Ob[(size_t)qRow * D_ + 16 * t + ln] = oacc[t][r] * inv[r];
    }
  }
}

extern "C" void kernel_launch(void* const* d_in, const int* in_sizes, int n_in,
                              void* d_out, int out_size, void* d_ws, size_t ws_size,
                              hipStream_t stream) {
  const float* Q = (const float*)d_in[0];
  const float* V = (const float*)d_in[1];
  float* Out = (float*)d_out;
  fa_fwd<<<dim3(B_ * (SQ_ / TQ)), dim3(NTHREADS), 0, stream>>>(Q, V, Out);
}
